// LSTMCell_8632884265137
// MI455X (gfx1250) — matrix operand order repaired and checked by reading, not compiled, not run on hardware
//
#include <hip/hip_runtime.h>

// ---------------------------------------------------------------------------
// Fused LSTM cell for gfx1250 (MI455X), split-bf16 WMMA GEMM + gating epilogue
//   ifgo = [x|h] @ w ; i,f,o = sigmoid ; g = tanh
//   c' = f*c + i*g ; h' = o*tanh(c') ; out = [h', h', c']
// B=1024, U=2048, K=4096, 4U=8192
//
// Pipeline per k-block (K=32):
//   regs(kb) --pack(v_perm)+ds_store_b32--> LDS | barrier |
//   prefetch regs(kb+1) from global         (in flight during...)
//   ds_load_b128 fragments + 48x v_wmma_f32_16x16x32_bf16 | barrier |
// ---------------------------------------------------------------------------

#define TILE_M 128      // rows per workgroup
#define TILE_N 64       // gate-columns per workgroup (x4 gates held at once)
#define TILE_K 32       // K per WMMA step (bf16 16x16x32)
#define KSTEPS 128      // 4096 / 32
#define LDS_STRIDE 40   // padded row stride (ushort elems), 80B = 5x16B

typedef __attribute__((ext_vector_type(16))) __bf16 v16bf;
typedef __attribute__((ext_vector_type(8)))  __bf16 v8bf;
typedef __attribute__((ext_vector_type(8)))  float  v8f;

// Pack two f32 into {bf16_hi(a1), bf16_hi(a0)} and the residual pair.
// hi = truncated-to-bf16 (exactly representable), r = a - hi exact in f32,
// so hi*hi + hi*lo + lo*hi recovers ~f32 product accuracy in the GEMM.
__device__ __forceinline__ void pack_pair(float a0, float a1,
                                          unsigned &hi, unsigned &lo) {
    unsigned u0 = __float_as_uint(a0);
    unsigned u1 = __float_as_uint(a1);
    hi = __builtin_amdgcn_perm(u1, u0, 0x07060302u);   // {u1[31:16], u0[31:16]}
    float r0 = a0 - __uint_as_float(u0 & 0xFFFF0000u);
    float r1 = a1 - __uint_as_float(u1 & 0xFFFF0000u);
    lo = __builtin_amdgcn_perm(__float_as_uint(r1), __float_as_uint(r0), 0x07060302u);
}

// Build a 16-elem bf16 fragment from two 16-byte LDS chunks.
__device__ __forceinline__ v16bf frag16(const unsigned short* p0, const unsigned short* p1) {
    union { v16bf v; v8bf h[2]; } u;
    u.h[0] = *(const v8bf*)p0;
    u.h[1] = *(const v8bf*)p1;
    return u.v;
}

__device__ __forceinline__ float sigmoid_f(float x) { return 1.0f / (1.0f + __expf(-x)); }
__device__ __forceinline__ float tanh_f(float x)    { return 2.0f / (1.0f + __expf(-2.0f * x)) - 1.0f; }

__global__ __launch_bounds__(256)
void lstm_cell_wmma(const float* __restrict__ x,
                    const float* __restrict__ hin,
                    const float* __restrict__ cin,
                    const float* __restrict__ w,
                    float* __restrict__ out)
{
    constexpr int U  = 2048;          // hidden size
    constexpr int N4 = 8192;          // 4U (ifgo width)
    constexpr int BU = 1024 * U;      // elements per output tensor

    // bf16 hi/lo staged tiles: A = 128x32, B(transposed) = 256x32 (4 gates x 64)
    __shared__ __align__(16) unsigned short sA_hi[TILE_M * LDS_STRIDE];
    __shared__ __align__(16) unsigned short sA_lo[TILE_M * LDS_STRIDE];
    __shared__ __align__(16) unsigned short sB_hi[4 * TILE_N * LDS_STRIDE];
    __shared__ __align__(16) unsigned short sB_lo[4 * TILE_N * LDS_STRIDE];

    const int tid  = threadIdx.x;
    const int lane = tid & 31;
    const int wv   = tid >> 5;     // wave id 0..7
    const int wm   = wv & 3;       // 4 row slots of 32
    const int wn   = wv >> 2;      // 2 col slots of 32
    const int m0   = blockIdx.y * TILE_M;   // global row base
    const int n0   = blockIdx.x * TILE_N;   // gate-column base (0..2047)

    // ---- staging thread mapping ----
    // A: thread owns k-pair kp (0,2,..,30) in rows mrow, mrow+16, ..., mrow+112
    const int kp   = (tid & 15) * 2;
    const int mrow = tid >> 4;              // 0..15
    // B: thread owns ifgo column cc over all 32 k's (pair-looped)
    const int cc   = tid;                                  // 0..255
    const int gcol = ((cc >> 6) * U) + n0 + (cc & 63);     // global ifgo column

    // ---- prefetch registers (raw f32 tile for next k-block) ----
    float2 pa[8];     // A: 8 rows x k-pair
    float2 pb[16];    // B: 16 k-pairs for column cc

    // Accumulators: [gate][mtile][ntile], each 16x16 f32 (8 VGPRs)
    v8f acc[4][2][2];
#pragma unroll
    for (int g = 0; g < 4; ++g)
#pragma unroll
        for (int mt = 0; mt < 2; ++mt)
#pragma unroll
            for (int nt = 0; nt < 2; ++nt)
                acc[g][mt][nt] = v8f{};

    // A-fragment lane decode (16-bit A 16x32 layout, ISA 7.12.2):
    //   lane<16 holds K 0..7 (v0..3) and 16..23 (v4..7); lane>=16: +8
    const int am  = lane & 15;
    const int g8  = (lane >> 4) << 3;   // 0 or 8
    // B-fragment lane decode (16-bit B 32x16): same K order as A (lane<16 K 0..7,16..23; lane>=16: +8)
    const int bn  = lane & 15;
    const int k16 = (lane >> 4) << 4;   // 0 or 16

    // ---------------- tile loader (global -> regs) ----------------
    auto load_tiles = [&](int kb) {
        const int k0 = kb * TILE_K;
        // concat(x, h): columns < U come from x, else from h
        const float* Asrc = (k0 < U) ? x : hin;
        const int    acol = (k0 < U) ? k0 : (k0 - U);
        const float* ap = Asrc + (size_t)m0 * U + acol + kp;
#pragma unroll
        for (int it = 0; it < 8; ++it)
            pa[it] = *(const float2*)(ap + (size_t)(mrow + it * 16) * U);
        const float* wp = w + (size_t)k0 * N4 + gcol;
#pragma unroll
        for (int kk = 0; kk < 16; ++kk) {
            pb[kk].x = wp[(size_t)(2 * kk)     * N4];
            pb[kk].y = wp[(size_t)(2 * kk + 1) * N4];
        }
    };

    // ---------------- tile converter (regs -> LDS bf16 hi/lo) ----------------
    auto store_tiles = [&]() {
#pragma unroll
        for (int it = 0; it < 8; ++it) {
            const int m = mrow + it * 16;
            unsigned hi, lo;
            pack_pair(pa[it].x, pa[it].y, hi, lo);
            *(unsigned*)&sA_hi[m * LDS_STRIDE + kp] = hi;
            *(unsigned*)&sA_lo[m * LDS_STRIDE + kp] = lo;
        }
#pragma unroll
        for (int kk = 0; kk < 16; ++kk) {
            unsigned hi, lo;
            pack_pair(pb[kk].x, pb[kk].y, hi, lo);
            *(unsigned*)&sB_hi[cc * LDS_STRIDE + 2 * kk] = hi;
            *(unsigned*)&sB_lo[cc * LDS_STRIDE + 2 * kk] = lo;
        }
    };

    // ---------------- software-pipelined main loop ----------------
    load_tiles(0);
    for (int kb = 0; kb < KSTEPS; ++kb) {
        store_tiles();          // convert prefetched tile, write LDS
        __syncthreads();
        if (kb + 1 < KSTEPS)
            load_tiles(kb + 1); // global loads in flight during the WMMAs below

        // ---- load A fragments (2 m-tiles, hi+lo) ----
        v16bf ah[2], al[2];
#pragma unroll
        for (int mt = 0; mt < 2; ++mt) {
            const unsigned short* rh = &sA_hi[(wm * 32 + mt * 16 + am) * LDS_STRIDE];
            const unsigned short* rl = &sA_lo[(wm * 32 + mt * 16 + am) * LDS_STRIDE];
            ah[mt] = frag16(rh + g8, rh + 16 + g8);
            al[mt] = frag16(rl + g8, rl + 16 + g8);
        }

        // ---- 4 gates x 2x2 tiles x 3 split terms = 48 WMMAs per wave/step ----
#pragma unroll
        for (int g = 0; g < 4; ++g) {
            v16bf bh[2], bl[2];
#pragma unroll
            for (int nt = 0; nt < 2; ++nt) {
                const int c2 = g * TILE_N + wn * 32 + nt * 16 + bn;
                bh[nt] = frag16(&sB_hi[c2 * LDS_STRIDE + g8], &sB_hi[c2 * LDS_STRIDE + 16 + g8]);
                bl[nt] = frag16(&sB_lo[c2 * LDS_STRIDE + g8], &sB_lo[c2 * LDS_STRIDE + 16 + g8]);
            }
#pragma unroll
            for (int mt = 0; mt < 2; ++mt)
#pragma unroll
                for (int nt = 0; nt < 2; ++nt) {
                    v8f t = acc[g][mt][nt];
                    t = __builtin_amdgcn_wmma_f32_16x16x32_bf16(false, ah[mt], false, bh[nt], (short)0, t, false, false);
                    t = __builtin_amdgcn_wmma_f32_16x16x32_bf16(false, ah[mt], false, bl[nt], (short)0, t, false, false);
                    t = __builtin_amdgcn_wmma_f32_16x16x32_bf16(false, al[mt], false, bh[nt], (short)0, t, false, false);
                    acc[g][mt][nt] = t;
                }
        }
        __syncthreads();
    }

    // ---- fused gating epilogue ----
    // C/D layout: VGPR r, lanes 0-15 -> M=r, lanes 16-31 -> M=8+r; N = lane&15
    const int roff = (lane >= 16) ? 8 : 0;
#pragma unroll
    for (int mt = 0; mt < 2; ++mt) {
#pragma unroll
        for (int nt = 0; nt < 2; ++nt) {
#pragma unroll
            for (int r = 0; r < 8; ++r) {
                const int row = m0 + wm * 32 + mt * 16 + roff + r;
                const int col = n0 + wn * 32 + nt * 16 + (lane & 15);   // 0..2047
                const size_t idx = (size_t)row * U + col;
                const float iv = sigmoid_f(acc[0][mt][nt][r]);
                const float fv = sigmoid_f(acc[1][mt][nt][r]);
                const float gv = tanh_f(acc[2][mt][nt][r]);
                const float ov = sigmoid_f(acc[3][mt][nt][r]);
                const float cn = fv * cin[idx] + iv * gv;
                const float hn = ov * tanh_f(cn);
                out[idx]                  = hn;   // h_new (output 0)
                out[BU + idx]             = hn;   // h_new (output 1)
                out[2 * (size_t)BU + idx] = cn;   // c_new (output 2)
            }
        }
    }
}

extern "C" void kernel_launch(void* const* d_in, const int* in_sizes, int n_in,
                              void* d_out, int out_size, void* d_ws, size_t ws_size,
                              hipStream_t stream) {
    (void)in_sizes; (void)n_in; (void)out_size; (void)d_ws; (void)ws_size;
    const float* x = (const float*)d_in[0];
    const float* h = (const float*)d_in[1];
    const float* c = (const float*)d_in[2];
    const float* w = (const float*)d_in[3];
    float* out = (float*)d_out;

    dim3 grid(2048 / TILE_N, 1024 / TILE_M);   // 32 x 8 = 256 workgroups
    dim3 block(256);                            // 8 waves (wave32)
    lstm_cell_wmma<<<grid, block, 0, stream>>>(x, h, c, w, out);
}
